// BiLSTM_74388833567105
// MI455X (gfx1250) — matrix operand order repaired and checked by reading, not compiled, not run on hardware
//
#include <hip/hip_runtime.h>
#include <hip/hip_bf16.h>
#include <math.h>

typedef __attribute__((ext_vector_type(16))) _Float16 v16h;
typedef __attribute__((ext_vector_type(8)))  float    v8f;

namespace {
constexpr int Bn  = 32;
constexpr int Tn  = 256;
constexpr int En  = 128;   // E_SZ
constexpr int Hn  = 64;    // H
constexpr int H2n = 128;   // 2H
constexpr int G4n = 256;   // 4H
constexpr int Un  = 64;    // U
constexpr int NEn = 256;   // NE
}

// ---------------------------------------------------------------------------
// WMMA fragment helpers (wave32, 16x16x32 f16 -> f32)
// ---------------------------------------------------------------------------
struct __align__(16) Q16 { unsigned int a, b, c, d; };
union H8 { Q16 q; _Float16 h[8]; };

// A fragment: rows m0..m0+15, K-slice k0..k0+31, A row-major (lda in halves).
// ISA layout: lanes 0-15 hold K {0..7,16..23}, lanes 16-31 hold K {8..15,24..31}.
__device__ __forceinline__ v16h load_a_frag(const _Float16* A, int lda, int m0, int k0) {
  const int lane = threadIdx.x & 31;
  const int r  = lane & 15;
  const int hi = lane >> 4;
  const _Float16* p = A + (size_t)(m0 + r) * lda + k0 + (hi << 3);
  H8 x0, x1;
  x0.q = *(const Q16*)(p);
  x1.q = *(const Q16*)(p + 16);
  v16h v;
#pragma unroll
  for (int i = 0; i < 8; ++i) { v[i] = x0.h[i]; v[8 + i] = x1.h[i]; }
  return v;
}

// B fragment from weight stored (N,K) row-major (ldw in halves): column n0+lane%16,
// lanes 0-15 hold K {0..7,16..23}, lanes 16-31 hold K {8..15,24..31}.
__device__ __forceinline__ v16h load_b_frag(const _Float16* W, int ldw, int n0, int k0) {
  const int lane = threadIdx.x & 31;
  const int c  = lane & 15;
  const int hi = lane >> 4;
  const _Float16* p = W + (size_t)(n0 + c) * ldw + k0 + (hi << 3);
  H8 x0, x1;
  x0.q = *(const Q16*)(p);
  x1.q = *(const Q16*)(p + 16);
  v16h v;
#pragma unroll
  for (int i = 0; i < 8; ++i) { v[i] = x0.h[i]; v[8 + i] = x1.h[i]; }
  return v;
}

// C/D fragment: VGPR r holds row m0 + r + (lane<16?0:8), column n0 + lane%16.
__device__ __forceinline__ void store_c_frag(float* C, int ldc, int m0, int n0, v8f acc) {
  const int lane = threadIdx.x & 31;
  const int c  = lane & 15;
  const int hi = lane >> 4;
#pragma unroll
  for (int r = 0; r < 8; ++r)
    C[(size_t)(m0 + r + hi * 8) * ldc + n0 + c] = acc[r];
}

__device__ __forceinline__ float sigm(float x) { return 1.f / (1.f + __expf(-x)); }

// Async global->LDS copy, 16B per lane (GLOBAL_LOAD_ASYNC_TO_LDS_B128, ASYNCcnt).
// Generic LDS addresses carry the LDS byte offset in bits [31:0] (aperture is in
// the high word), so truncation yields the VDST LDS address the ISA expects.
__device__ __forceinline__ void async_copy_b128(const void* gsrc, void* ldst) {
  unsigned int loff = (unsigned int)(uintptr_t)ldst;
  asm volatile("global_load_async_to_lds_b128 %0, %1, off"
               :: "v"(loff), "v"(gsrc) : "memory");
}
__device__ __forceinline__ void wait_asynccnt_le4() {
  asm volatile("s_wait_asynccnt 0x4" ::: "memory");
}

// ---------------------------------------------------------------------------
// K1: f32 -> f16 cast (weights)
// ---------------------------------------------------------------------------
__global__ void cast_f32_f16(const float* __restrict__ in, _Float16* __restrict__ out, int n) {
  int i = blockIdx.x * blockDim.x + threadIdx.x;
  if (i < n) out[i] = (_Float16)in[i];
}

// K2: embedding gather -> f16 activations
__global__ void gather_embed(const int* __restrict__ ids, const float* __restrict__ emb,
                             _Float16* __restrict__ out, int total) {
  int i = blockIdx.x * blockDim.x + threadIdx.x;
  if (i < total) {
    int row = i >> 7;          // /E (=128)
    int col = i & 127;
    out[i] = (_Float16)emb[(size_t)ids[row] * En + col];
  }
}

// ---------------------------------------------------------------------------
// K3: generic WMMA GEMM  C(M,N) = A(M,K) @ W(N,K)^T (+ bias[N])
// One 16x16 tile per wave, 8 waves per block. M,N multiples of 16, K of 32.
// ---------------------------------------------------------------------------
__global__ __launch_bounds__(256)
void gemm_wmma_f16(const _Float16* __restrict__ A, const _Float16* __restrict__ W,
                   const float* __restrict__ bias, float* __restrict__ C,
                   int M, int N, int K) {
  const int gwave = (blockIdx.x * blockDim.x + threadIdx.x) >> 5;
  const int ntiles = N >> 4;
  const int mt = gwave / ntiles;
  const int nt = gwave - mt * ntiles;
  v8f acc = {};
  for (int k0 = 0; k0 < K; k0 += 32) {
    v16h a = load_a_frag(A, K, mt * 16, k0);
    v16h b = load_b_frag(W, K, nt * 16, k0);
    acc = __builtin_amdgcn_wmma_f32_16x16x32_f16(false, a, false, b,
                                                 (short)0, acc, false, false);
  }
  if (bias) {
    float bv = bias[nt * 16 + (threadIdx.x & 15)];
#pragma unroll
    for (int r = 0; r < 8; ++r) acc[r] += bv;
  }
  store_c_frag(C, N, mt * 16, nt * 16, acc);
}

// ---------------------------------------------------------------------------
// K4: persistent LSTM recurrence, one block per (lstm, direction).
// State in LDS; Wh fragments preloaded in VGPRs; WMMA h@Wh.T each step.
// Xproj slice for step t+1 streamed into a double-buffered LDS staging area
// with GLOBAL_LOAD_ASYNC_TO_LDS_B128, overlapped with step t compute.
// ---------------------------------------------------------------------------
__global__ __launch_bounds__(512)
void lstm_seq_wmma(const float* __restrict__ xpAll,   // 4 contiguous (B,T,4H) f32
                   const _Float16* __restrict__ whAll,// 4 contiguous (4H,H) f16
                   _Float16* __restrict__ outS,       // (B,T,H2) sentence
                   _Float16* __restrict__ outD) {     // (B,T,H2) data
  const int blk = blockIdx.x;                 // 0:S-fwd 1:S-bwd 2:D-fwd 3:D-bwd
  const float*    xproj = xpAll + (size_t)blk * Bn * Tn * G4n;
  const _Float16* Wh    = whAll + (size_t)blk * G4n * Hn;
  _Float16*       out   = ((blk < 2) ? outS : outD) + (blk & 1) * Hn;
  const int reverse = blk & 1;

  __shared__ __align__(16) _Float16 hS[Bn * Hn];     // 4 KB
  __shared__ float cS[Bn * Hn];                      // 8 KB
  __shared__ float zS[Bn * G4n];                     // 32 KB
  __shared__ __align__(16) float xbuf[2][Bn * G4n];  // 2 x 32 KB staging

  const int tid  = threadIdx.x;
  const int wave = tid >> 5;                         // 16 waves; wave w owns N-tile w

  for (int i = tid; i < Bn * Hn; i += 512) { hS[i] = (_Float16)0.f; cS[i] = 0.f; }

  // Preload B fragments for this wave's N-tile (invariant over time steps).
  v16h bf0 = load_b_frag(Wh, Hn, wave * 16, 0);
  v16h bf1 = load_b_frag(Wh, Hn, wave * 16, 32);

  // Kick off async staging of the step-0 Xproj slice (4 x B128 per thread).
  {
    const int tt0 = reverse ? (Tn - 1) : 0;
#pragma unroll
    for (int i = 0; i < 4; ++i) {
      const int c = i * 512 + tid;          // 16-byte chunk id, 0..2047
      const int b = c >> 6;                 // batch row
      const int n = (c & 63) * 4;           // gate column
      async_copy_b128(xproj + ((size_t)b * Tn + tt0) * G4n + n,
                      &xbuf[0][b * G4n + n]);
    }
  }
  __syncthreads();

  for (int t = 0; t < Tn; ++t) {
    const int p = t & 1;

    // Issue async copy for step t+1 into the other buffer (free since the
    // barrier at the end of step t-1 retired all readers of it).
    {
      const int tnext = (t + 1 < Tn) ? (t + 1) : t;
      const int ttn = reverse ? (Tn - 1 - tnext) : tnext;
#pragma unroll
      for (int i = 0; i < 4; ++i) {
        const int c = i * 512 + tid;
        const int b = c >> 6;
        const int n = (c & 63) * 4;
        async_copy_b128(xproj + ((size_t)b * Tn + ttn) * G4n + n,
                        &xbuf[p ^ 1][b * G4n + n]);
      }
    }

    // z_rec = h @ Wh.T : M=32 (2 tiles) x N-tile = 4 WMMAs per wave
    v8f acc0 = {}, acc1 = {};
    {
      v16h a00 = load_a_frag(hS, Hn, 0, 0);
      v16h a10 = load_a_frag(hS, Hn, 16, 0);
      acc0 = __builtin_amdgcn_wmma_f32_16x16x32_f16(false, a00, false, bf0,
                                                    (short)0, acc0, false, false);
      acc1 = __builtin_amdgcn_wmma_f32_16x16x32_f16(false, a10, false, bf0,
                                                    (short)0, acc1, false, false);
      v16h a01 = load_a_frag(hS, Hn, 0, 32);
      v16h a11 = load_a_frag(hS, Hn, 16, 32);
      acc0 = __builtin_amdgcn_wmma_f32_16x16x32_f16(false, a01, false, bf1,
                                                    (short)0, acc0, false, false);
      acc1 = __builtin_amdgcn_wmma_f32_16x16x32_f16(false, a11, false, bf1,
                                                    (short)0, acc1, false, false);
    }
    store_c_frag(zS, G4n, 0,  wave * 16, acc0);
    store_c_frag(zS, G4n, 16, wave * 16, acc1);

    // Retire this step's 4 async copies (loads complete in order: ASYNCcnt<=4
    // leaves only the step-t+1 copies in flight), then make LDS visible.
    wait_asynccnt_le4();
    __syncthreads();

    // Gate math: 32*64 = 2048 (b,j) cells, 4 per thread. All operands in LDS.
    const int tt = reverse ? (Tn - 1 - t) : t;
    for (int e = tid; e < Bn * Hn; e += 512) {
      const int b = e >> 6;
      const int j = e & 63;
      const float* xp = &xbuf[p][b * G4n];
      const float* zr = zS + b * G4n;
      float zi = zr[j]           + xp[j];
      float zf = zr[Hn + j]      + xp[Hn + j];
      float zg = zr[2 * Hn + j]  + xp[2 * Hn + j];
      float zo = zr[3 * Hn + j]  + xp[3 * Hn + j];
      float c  = sigm(zf) * cS[e] + sigm(zi) * tanhf(zg);
      float h  = sigm(zo) * tanhf(c);
      cS[e] = c;
      hS[e] = (_Float16)h;
      out[((size_t)b * Tn + tt) * H2n + j] = (_Float16)h;
    }
    __syncthreads();
  }
}

// ---------------------------------------------------------------------------
// K5: entity segment means (per-batch scan, channel-parallel accumulation)
// ---------------------------------------------------------------------------
__global__ __launch_bounds__(128)
void getentity_kernel(const int* __restrict__ labels, const _Float16* __restrict__ seqout,
                      _Float16* __restrict__ ent) {
  const int b   = blockIdx.x;
  const int ch  = threadIdx.x;   // 0..127
  __shared__ int stS[Tn];
  __shared__ int eidS[Tn];
  __shared__ int closedS;
  if (ch == 0) {
    int s = 0, e = -1, nstart = 0;
    for (int t = 0; t < Tn; ++t) {
      int l = labels[b * Tn + t];
      if (l == 1)      { s = 1; e = nstart; ++nstart; }
      else if (l == 0) { s = 0; }
      stS[t] = s; eidS[t] = e;
    }
    closedS = nstart - s;        // drop trailing open entity
  }
  __syncthreads();
  const int closed = closedS;

  for (int e = 0; e < NEn; ++e)
    ent[((size_t)b * NEn + e) * H2n + ch] = (_Float16)0.f;

  float acc = 0.f; int cnt = 0; int cur = -1;
  for (int t = 0; t < Tn; ++t) {
    if (stS[t]) {
      if (eidS[t] != cur) { acc = 0.f; cnt = 0; cur = eidS[t]; }
      acc += (float)seqout[((size_t)b * Tn + t) * H2n + ch];
      ++cnt;
      bool ends = (t == Tn - 1) || (!stS[t + 1]) || (eidS[t + 1] != cur);
      if (ends && cur < closed)
        ent[((size_t)b * NEn + cur) * H2n + ch] = (_Float16)(acc / (float)cnt);
    }
  }
}

// ---------------------------------------------------------------------------
// K6: logits + softmax over d.  One block per (b,w) row.
// ---------------------------------------------------------------------------
__global__ __launch_bounds__(256)
void attn_softmax(const float* __restrict__ enc, const float* __restrict__ dec,
                  const float* __restrict__ v, float* __restrict__ out) {
  const int bw = blockIdx.x;           // b*NE + w
  const int b  = bw >> 8;
  const int d  = threadIdx.x;          // 0..255
  __shared__ float decL[Un];
  __shared__ float vL[Un];
  __shared__ float red[NEn];
  if (d < Un) { decL[d] = dec[(size_t)bw * Un + d]; vL[d] = v[d]; }
  __syncthreads();

  const float* er = enc + ((size_t)b * NEn + d) * Un;
  float s = 0.f;
#pragma unroll 8
  for (int u = 0; u < Un; ++u) s += vL[u] * tanhf(er[u] + decL[u]);

  red[d] = s; __syncthreads();
  for (int off = 128; off > 0; off >>= 1) {
    if (d < off) red[d] = fmaxf(red[d], red[d + off]);
    __syncthreads();
  }
  const float m = red[0];
  __syncthreads();
  const float ex = __expf(s - m);
  red[d] = ex; __syncthreads();
  for (int off = 128; off > 0; off >>= 1) {
    if (d < off) red[d] += red[d + off];
    __syncthreads();
  }
  out[(size_t)bw * NEn + d] = ex / red[0];
}

// ---------------------------------------------------------------------------
// Host-side orchestration
// ---------------------------------------------------------------------------
extern "C" void kernel_launch(void* const* d_in, const int* in_sizes, int n_in,
                              void* d_out, int out_size, void* d_ws, size_t ws_size,
                              hipStream_t stream) {
  (void)in_sizes; (void)n_in; (void)out_size; (void)ws_size;

  const int*   sents    = (const int*)d_in[0];
  const int*   datas    = (const int*)d_in[1];
  const int*   slabels  = (const int*)d_in[2];
  const int*   dlabels  = (const int*)d_in[3];
  const float* emb      = (const float*)d_in[6];
  const float* Wi[4]    = { (const float*)d_in[7],  (const float*)d_in[10],
                            (const float*)d_in[13], (const float*)d_in[16] };
  const float* Whf[4]   = { (const float*)d_in[8],  (const float*)d_in[11],
                            (const float*)d_in[14], (const float*)d_in[17] };
  const float* bias[4]  = { (const float*)d_in[9],  (const float*)d_in[12],
                            (const float*)d_in[15], (const float*)d_in[18] };
  const float* W1       = (const float*)d_in[19];
  const float* W2       = (const float*)d_in[20];
  const float* vvec     = (const float*)d_in[21];
  float* out            = (float*)d_out;

  // Workspace carving (256-byte aligned)
  char* base = (char*)d_ws;
  size_t off = 0;
  auto carve = [&](size_t bytes) -> char* {
    off = (off + 255) & ~(size_t)255;
    char* p = base + off;
    off += bytes;
    return p;
  };
  _Float16* esH   = (_Float16*)carve((size_t)Bn * Tn * En * 2);
  _Float16* edH   = (_Float16*)carve((size_t)Bn * Tn * En * 2);
  _Float16* wiH   = (_Float16*)carve((size_t)4 * G4n * En * 2);
  _Float16* whH   = (_Float16*)carve((size_t)4 * G4n * Hn * 2);
  _Float16* w1H   = (_Float16*)carve((size_t)Un * H2n * 2);
  _Float16* w2H   = (_Float16*)carve((size_t)Un * H2n * 2);
  float*    xpAll = (float*)carve((size_t)4 * Bn * Tn * G4n * 4);
  _Float16* rnnH  = (_Float16*)carve((size_t)Bn * Tn * H2n * 2);
  _Float16* dataH = (_Float16*)carve((size_t)Bn * Tn * H2n * 2);
  _Float16* wentH = (_Float16*)carve((size_t)Bn * NEn * H2n * 2);
  _Float16* dentH = (_Float16*)carve((size_t)Bn * NEn * H2n * 2);
  float*    encF  = (float*)carve((size_t)Bn * NEn * Un * 4);
  float*    decF  = (float*)carve((size_t)Bn * NEn * Un * 4);

  // 1. Weight casts to f16
  for (int i = 0; i < 4; ++i) {
    cast_f32_f16<<<(G4n * En + 255) / 256, 256, 0, stream>>>(Wi[i],  wiH + (size_t)i * G4n * En, G4n * En);
    cast_f32_f16<<<(G4n * Hn + 255) / 256, 256, 0, stream>>>(Whf[i], whH + (size_t)i * G4n * Hn, G4n * Hn);
  }
  cast_f32_f16<<<(Un * H2n + 255) / 256, 256, 0, stream>>>(W1, w1H, Un * H2n);
  cast_f32_f16<<<(Un * H2n + 255) / 256, 256, 0, stream>>>(W2, w2H, Un * H2n);

  // 2. Embedding gathers
  {
    const int total = Bn * Tn * En;
    gather_embed<<<(total + 255) / 256, 256, 0, stream>>>(sents, emb, esH, total);
    gather_embed<<<(total + 255) / 256, 256, 0, stream>>>(datas, emb, edH, total);
  }

  // 3. Input projections: Xproj = es/ed @ Wi.T + b  (M=8192, N=256, K=128)
  {
    const int M = Bn * Tn, N = G4n, K = En;
    const int blocks = (M / 16) * (N / 16) / 8;
    const _Float16* Ain[4] = { esH, esH, edH, edH };
    for (int i = 0; i < 4; ++i)
      gemm_wmma_f16<<<blocks, 256, 0, stream>>>(
          Ain[i], wiH + (size_t)i * G4n * En, bias[i],
          xpAll + (size_t)i * Bn * Tn * G4n, M, N, K);
  }

  // 4. Recurrences: 4 persistent workgroups
  lstm_seq_wmma<<<4, 512, 0, stream>>>(xpAll, whH, rnnH, dataH);

  // 5. Entity segment means
  getentity_kernel<<<Bn, 128, 0, stream>>>(dlabels, dataH, dentH);
  getentity_kernel<<<Bn, 128, 0, stream>>>(slabels, rnnH, wentH);

  // 6. enc = dataent @ W1.T ; dec = wordent @ W2.T  (M=8192, N=64, K=128)
  {
    const int M = Bn * NEn, N = Un, K = H2n;
    const int blocks = (M / 16) * (N / 16) / 8;
    gemm_wmma_f16<<<blocks, 256, 0, stream>>>(dentH, w1H, nullptr, encF, M, N, K);
    gemm_wmma_f16<<<blocks, 256, 0, stream>>>(wentH, w2H, nullptr, decF, M, N, K);
  }

  // 7. Attention logits + softmax over d
  attn_softmax<<<Bn * NEn, 256, 0, stream>>>(encF, decF, vvec, out);
}
